// RetrievalModel_79319456023259
// MI455X (gfx1250) — compile-verified, matrix operand order checked by reading, not run on hardware
//
#include <hip/hip_runtime.h>
#include <hip/hip_bf16.h>
#include <stdint.h>

typedef __attribute__((ext_vector_type(16))) __bf16 v16bf;
typedef __attribute__((ext_vector_type(8)))  float  v8f;

#define NQ          256        // queries
#define DIM         64         // feature dim
#define K_TOP       100        // top-k
#define P_NORM      0.495f     // 0.5 * 0.99, applied to squared norms
#define GBLOCKS     256        // partition blocks for main kernel
#define SCORE_PITCH 17         // 16 + 1 pad to avoid LDS bank conflicts

// ---------- order-preserving float <-> u32 key transform ----------
__device__ __forceinline__ uint32_t f2ord(float f) {
    uint32_t b = __float_as_uint(f);
    return b ^ ((b & 0x80000000u) ? 0xFFFFFFFFu : 0x80000000u);
}
__device__ __forceinline__ float ord2f(uint32_t u) {
    uint32_t b = (u & 0x80000000u) ? (u ^ 0x80000000u) : ~u;
    return __uint_as_float(b);
}

// ---------- load one 16-half WMMA fragment (split fp32 -> bf16 hi/lo) ----------
// 16-bit A/B fragment layout (ISA 7.12.2): lanes 0-15 hold K in {0..7, 16..23},
// lanes 16-31 hold K in {8..15, 24..31} (relative to 32-wide K-chunk kc).
__device__ __forceinline__ void load_frag_split(const float* __restrict__ rowPtr,
                                                int kc, int laneHalf,
                                                v16bf& hi, v16bf& lo) {
    const float* p0 = rowPtr + kc * 32 + laneHalf * 8;
    const float* p1 = p0 + 16;
    float f[16];
#pragma unroll
    for (int i = 0; i < 8; ++i) f[i] = p0[i];
#pragma unroll
    for (int i = 0; i < 8; ++i) f[8 + i] = p1[i];
#pragma unroll
    for (int i = 0; i < 16; ++i) {
        __bf16 h = (__bf16)f[i];
        hi[i] = h;
        lo[i] = (__bf16)(f[i] - (float)h);
    }
}

// ---------- min-heap sift (interleaved LDS layout: slot(i) = heap[i*NQ + t]) ----------
__device__ __forceinline__ void heap_sift(unsigned long long* __restrict__ heap,
                                          int t, unsigned long long key, int size) {
    int pos = 0;
    for (;;) {
        int cl = 2 * pos + 1;
        if (cl >= size) break;
        int cr = cl + 1;
        int mc = (cr < size && heap[cr * NQ + t] < heap[cl * NQ + t]) ? cr : cl;
        if (heap[mc * NQ + t] >= key) break;
        heap[pos * NQ + t] = heap[mc * NQ + t];
        pos = mc;
    }
    heap[pos * NQ + t] = key;
}

// ---------- kernel 1: inverse-power squared-norm scales ----------
__global__ __launch_bounds__(256) void invnorm_kernel(const float* __restrict__ X,
                                                      float* __restrict__ invn, int n) {
    int i = blockIdx.x * blockDim.x + threadIdx.x;
    if (i >= n) return;
    const float4* r = (const float4*)(X + (size_t)i * DIM);
    float s = 0.f;
#pragma unroll
    for (int j = 0; j < DIM / 4; ++j) {
        float4 v = r[j];
        s += v.x * v.x + v.y * v.y + v.z * v.z + v.w * v.w;
    }
    s = fmaxf(s, 1e-30f);
    invn[i] = exp2f(-P_NORM * log2f(s));   // s^(-0.495)
}

// ---------- kernel 2: fused split-bf16 WMMA scoring + per-block top-k ----------
__global__ __launch_bounds__(256) void score_topk_kernel(const float* __restrict__ Q,
                                                         const float* __restrict__ C,
                                                         const float* __restrict__ inv_cn,
                                                         unsigned long long* __restrict__ parts,
                                                         int N) {
    extern __shared__ char smem[];
    unsigned long long* heap = (unsigned long long*)smem;                  // NQ * K_TOP u64
    float* scorebuf = (float*)(smem + (size_t)NQ * K_TOP * 8);             // [NQ][SCORE_PITCH]

    const int t     = threadIdx.x;
    const int lane  = t & 31;
    const int wave  = t >> 5;
    const int nhalf = lane >> 4;   // selects K sub-band per fragment layout
    const int nidx  = lane & 15;   // M (for A) / N (for B) within the 16-wide tile

    // init heaps: key 0 == -FLT_MAX sentinel
    for (int j = 0; j < K_TOP; ++j) heap[j * NQ + t] = 0ull;

    // A fragments resident in VGPRs: wave handles query tiles 2*wave, 2*wave+1
    v16bf a_hi[2][2], a_lo[2][2];
#pragma unroll
    for (int qi = 0; qi < 2; ++qi) {
        const float* qrow = Q + (size_t)((2 * wave + qi) * 16 + nidx) * DIM;
#pragma unroll
        for (int kc = 0; kc < 2; ++kc)
            load_frag_split(qrow, kc, nhalf, a_hi[qi][kc], a_lo[qi][kc]);
    }
    __syncthreads();

    const int per    = (N + GBLOCKS - 1) / GBLOCKS;
    const int start  = blockIdx.x * per;
    const int end    = min(start + per, N);
    const int ntiles = (end - start + 15) / 16;

    for (int tile = 0; tile < ntiles; ++tile) {
        const int cbase = start + tile * 16;
        const int csafe = min(cbase + nidx, N - 1);
        const float* crow = C + (size_t)csafe * DIM;

        // Software prefetch of the NEXT candidate tile: this stream is the
        // HBM-latency-critical path; global_prefetch_b8 pulls whole cachelines
        // (each lane covers both 128B halves of its next-tile row -> full 4KB tile).
        {
            const int pnext = min(cbase + 16 + nidx, N - 1);
            const char* prow = (const char*)(C + (size_t)pnext * DIM);
            __builtin_prefetch(prow, 0, 3);
            __builtin_prefetch(prow + 128, 0, 3);
        }

        v16bf b_hi[2], b_lo[2];
#pragma unroll
        for (int kc = 0; kc < 2; ++kc)
            load_frag_split(crow, kc, nhalf, b_hi[kc], b_lo[kc]);
        const float sc = inv_cn[csafe];

#pragma unroll
        for (int qi = 0; qi < 2; ++qi) {
            v8f acc = {};
#pragma unroll
            for (int kc = 0; kc < 2; ++kc) {
                acc = __builtin_amdgcn_wmma_f32_16x16x32_bf16(
                    false, a_hi[qi][kc], false, b_hi[kc], (short)0, acc, false, false);
                acc = __builtin_amdgcn_wmma_f32_16x16x32_bf16(
                    false, a_hi[qi][kc], false, b_lo[kc], (short)0, acc, false, false);
                acc = __builtin_amdgcn_wmma_f32_16x16x32_bf16(
                    false, a_lo[qi][kc], false, b_hi[kc], (short)0, acc, false, false);
            }
            const int qt = 2 * wave + qi;
#pragma unroll
            for (int g = 0; g < 8; ++g) {
                int qrow = qt * 16 + g + 8 * nhalf;     // C/D layout: VGPR g -> M = g (+8 hi half)
                scorebuf[qrow * SCORE_PITCH + nidx] = acc[g] * sc;
            }
        }
        __syncthreads();

        // thread t owns query t: fold the 16 new scores into its top-k heap
        for (int j = 0; j < 16; ++j) {
            int cand = cbase + j;
            if (cand < end) {
                float v = scorebuf[t * SCORE_PITCH + j];
                unsigned long long key =
                    ((unsigned long long)f2ord(v) << 32) | (uint32_t)cand;
                if (key > heap[0 * NQ + t]) heap_sift(heap, t, key, K_TOP);
            }
        }
        __syncthreads();
    }

    // in-place heapsort -> array descending (index 0 = max)
    for (int size = K_TOP; size > 1; --size) {
        unsigned long long root = heap[0 * NQ + t];
        unsigned long long lastv = heap[(size - 1) * NQ + t];
        heap[(size - 1) * NQ + t] = root;
        heap_sift(heap, t, lastv, size - 1);
    }
    for (int j = 0; j < K_TOP; ++j)
        parts[((size_t)blockIdx.x * NQ + t) * K_TOP + j] = heap[j * NQ + t];
}

// ---------- kernel 3: per-query tournament merge of GBLOCKS sorted lists ----------
__global__ __launch_bounds__(256) void merge_topk_kernel(const unsigned long long* __restrict__ parts,
                                                         const float* __restrict__ inv_qn,
                                                         float* __restrict__ out_vals,
                                                         float* __restrict__ out_idx) {
    __shared__ unsigned long long skey[256];
    __shared__ int sown[256];
    const int q = blockIdx.x;
    const int t = threadIdx.x;
    const unsigned long long* list = parts + ((size_t)t * NQ + q) * K_TOP;
    const float iq = inv_qn[q];
    int cur = 0;
    for (int i = 0; i < K_TOP; ++i) {
        skey[t] = (cur < K_TOP) ? list[cur] : 0ull;
        sown[t] = t;
        __syncthreads();
        for (int s = 128; s > 0; s >>= 1) {
            if (t < s && skey[t + s] > skey[t]) {
                skey[t] = skey[t + s];
                sown[t] = sown[t + s];
            }
            __syncthreads();
        }
        int winner = sown[0];
        unsigned long long wkey = skey[0];
        __syncthreads();
        if (t == winner) ++cur;
        if (t == 0) {
            out_vals[q * K_TOP + i] = ord2f((uint32_t)(wkey >> 32)) * iq;
            out_idx[q * K_TOP + i]  = (float)(uint32_t)(wkey & 0xFFFFFFFFu);
        }
    }
}

extern "C" void kernel_launch(void* const* d_in, const int* in_sizes, int n_in,
                              void* d_out, int out_size, void* d_ws, size_t ws_size,
                              hipStream_t stream) {
    const float* Q = (const float*)d_in[0];           // [256, 64]
    const float* C = (const float*)d_in[1];           // [N, 64]
    const int N = in_sizes[1] / DIM;                  // 1,000,000

    // workspace layout
    float* inv_cn = (float*)d_ws;                                   // N floats
    float* inv_qn = inv_cn + (((size_t)N + 255) & ~(size_t)255);    // 256 floats
    unsigned long long* parts = (unsigned long long*)(inv_qn + 256);// GBLOCKS*NQ*K_TOP u64

    float* out_vals = (float*)d_out;
    float* out_idx  = out_vals + NQ * K_TOP;

    invnorm_kernel<<<(N + 255) / 256, 256, 0, stream>>>(C, inv_cn, N);
    invnorm_kernel<<<1, 256, 0, stream>>>(Q, inv_qn, NQ);

    const size_t smem = (size_t)NQ * K_TOP * 8 + (size_t)NQ * SCORE_PITCH * 4; // 222,208 B
    (void)hipFuncSetAttribute(reinterpret_cast<const void*>(score_topk_kernel),
                              hipFuncAttributeMaxDynamicSharedMemorySize, (int)smem);
    score_topk_kernel<<<GBLOCKS, 256, smem, stream>>>(Q, C, inv_cn, parts, N);

    merge_topk_kernel<<<NQ, 256, 0, stream>>>(parts, inv_qn, out_vals, out_idx);
}
